// MultiheadAttention_2448131358735
// MI455X (gfx1250) — matrix operand order repaired and checked by reading, not compiled, not run on hardware
//
#include <hip/hip_runtime.h>
#include <hip/hip_bf16.h>

#define SEQ 4096
#define DM 512
#define NH 8
#define DH 64

typedef __attribute__((ext_vector_type(16))) _Float16 v16h;
typedef __attribute__((ext_vector_type(8)))  _Float16 v8h;
typedef __attribute__((ext_vector_type(2)))  _Float16 h2;
typedef __attribute__((ext_vector_type(8)))  float    v8f;

__device__ __forceinline__ v8f wmma16(v16h a, v16h b, v8f c) {
  // v_wmma_f32_16x16x32_f16: (neg_a, A, neg_b, B, c_mod, C, reuse_a, reuse_b)
  return __builtin_amdgcn_wmma_f32_16x16x32_f16(false, a, false, b, (short)0, c, false, false);
}

__device__ __forceinline__ h2 pkmax(h2 a, h2 b) {
  h2 d;
  asm("v_pk_max_num_f16 %0, %1, %2" : "=v"(d) : "v"(a), "v"(b));
  return d;
}

__device__ __forceinline__ h2 shfl_xor_h2(h2 a, int mask) {
  int ai = __builtin_bit_cast(int, a);
  int bi = __shfl_xor(ai, mask, 32);
  return __builtin_bit_cast(h2, bi);
}

// A-matrix 16x32 f16 (MxK), rows row-major in memory with leading dim `ld`.
// ISA layout: lane<16 holds M=lane, K={0..7, 16..23}; lane>=16 holds K={8..15, 24..31}.
__device__ __forceinline__ v16h load_a16(const _Float16* base, int ld, int lane) {
  const _Float16* rp = base + (size_t)(lane & 15) * ld + ((lane >> 4) << 3);
  v8h lo = *(const v8h*)rp;         // K = sel*8 .. +7
  v8h hi = *(const v8h*)(rp + 16);  // K = 16 + sel*8 .. +7
  v16h a;
#pragma unroll
  for (int i = 0; i < 8; ++i) { a[i] = lo[i]; a[i + 8] = hi[i]; }
  return a;
}

// B-matrix 32x16 f16 (KxN), stored column-contiguous in K: element (K,N) at base + N*ld + K.
// ISA layout: lane holds N=lane%16, K={0..7, 16..23} for lane<16; K={8..15, 24..31} for lane>=16.
__device__ __forceinline__ v16h load_b16(const _Float16* base, int ld, int lane) {
  const _Float16* p = base + (size_t)(lane & 15) * ld + ((lane >> 4) << 3);
  v8h lo = *(const v8h*)p;
  v8h hi = *(const v8h*)(p + 16);
  return __builtin_shufflevector(lo, hi, 0, 1, 2, 3, 4, 5, 6, 7, 8, 9, 10, 11, 12, 13, 14, 15);
}

// ---------------- prep kernels ----------------

__global__ void k_cvt(const float* __restrict__ in, _Float16* __restrict__ out, int n) {
  int i = blockIdx.x * blockDim.x + threadIdx.x;
  if (i < n) out[i] = (_Float16)in[i];
}

// in: [NH][DM][DH] f32 -> out: [NH][DH][DM] f16
__global__ void k_transpose_w(const float* __restrict__ in, _Float16* __restrict__ out) {
  int i = blockIdx.x * blockDim.x + threadIdx.x;  // NH*DM*DH = 262144
  int h  = i >> 15;
  int r  = i & 32767;
  int d  = r >> 6;
  int dh = r & 63;
  out[((size_t)h * DH + dh) * DM + d] = (_Float16)in[i];
}

// ---------------- QKV projection ----------------
// WT: [3][NH][DH][DM] f16 (transposed weights). q,k: [NH][SEQ][DH] f16. vT: [NH][DH][SEQ] f16.
__global__ void __launch_bounds__(256) k_qkv(const _Float16* __restrict__ xh,
                                             const _Float16* __restrict__ WT,
                                             const float* __restrict__ bq,
                                             const float* __restrict__ bk,
                                             const float* __restrict__ bv,
                                             _Float16* __restrict__ q,
                                             _Float16* __restrict__ k,
                                             _Float16* __restrict__ vT) {
  int lane = threadIdx.x & 31;
  int wid  = blockIdx.x * (blockDim.x >> 5) + (threadIdx.x >> 5);  // 0..24575
  int s_tile = wid / 96;
  int rem    = wid % 96;
  int mat_h  = rem >> 2;   // mat*8 + h
  int ntile  = rem & 3;
  int mat = mat_h >> 3;
  int h   = mat_h & 7;
  int m0 = s_tile * 16;
  int n0 = ntile * 16;
  const _Float16* W = WT + ((size_t)mat * NH + h) * DH * DM;

  v8f c = {};
#pragma unroll 4
  for (int kk = 0; kk < DM; kk += 32) {
    v16h a = load_a16(xh + (size_t)m0 * DM + kk, DM, lane);
    v16h b = load_b16(W + (size_t)n0 * DM + kk, DM, lane);
    c = wmma16(a, b, c);
  }
  const float* bias = (mat == 0) ? bq : (mat == 1) ? bk : bv;
  int   n  = n0 + (lane & 15);
  float bn = bias[h * DH + n];
  int   mb = m0 + ((lane >> 4) << 3);
  if (mat == 2) {
    _Float16* dst = vT + ((size_t)h * DH + n) * SEQ;
#pragma unroll
    for (int r = 0; r < 8; ++r) dst[mb + r] = (_Float16)(c[r] + bn);
  } else {
    const float scale = (mat == 0) ? 0.125f : 1.0f;  // fold 1/sqrt(DH) into q
    _Float16* dst = (mat == 0 ? q : k) + (size_t)h * SEQ * DH;
#pragma unroll
    for (int r = 0; r < 8; ++r)
      dst[(size_t)(mb + r) * DH + n] = (_Float16)((c[r] + bn) * scale);
  }
}

// ---------------- flash attention ----------------
// 8 waves/block on the SAME head, lockstepped per t-chunk so K/V tiles hit WGP$.
// Row sums come free from an extra ones-matmul accumulator (no sum shuffles).
__global__ void __launch_bounds__(256) k_attn(const _Float16* __restrict__ q,
                                              const _Float16* __restrict__ k,
                                              const _Float16* __restrict__ vT,
                                              _Float16* __restrict__ ctx) {
  __shared__ _Float16 pbuf[8][16 * 32];
  int lane = threadIdx.x & 31;
  int w    = threadIdx.x >> 5;
  int h    = blockIdx.y;
  int s0   = (blockIdx.x * 8 + w) * 16;
  int row  = lane & 15;
  int sel  = lane >> 4;
  int mb   = sel << 3;

  const _Float16* qh = q  + (size_t)h * SEQ * DH;
  const _Float16* kh = k  + (size_t)h * SEQ * DH;
  const _Float16* vh = vT + (size_t)h * DH * SEQ;

  v16h A0 = load_a16(qh + (size_t)s0 * DH + 0,  DH, lane);
  v16h A1 = load_a16(qh + (size_t)s0 * DH + 32, DH, lane);

  v16h Bones;
#pragma unroll
  for (int i = 0; i < 16; ++i) Bones[i] = (_Float16)1.0f;

  float m_i[8];
  v8f O0 = {}, O1 = {}, O2 = {}, O3 = {}, O4 = {};  // O4 = running row sums (ones-matmul)
#pragma unroll
  for (int r = 0; r < 8; ++r) m_i[r] = -1e30f;

  for (int t0 = 0; t0 < SEQ; t0 += 32) {
    __syncthreads();  // lockstep the 8 waves so K/V tiles are reused out of WGP$
    v16h B00 = load_b16(kh + (size_t)t0 * DH + 0,         DH, lane);
    v16h B01 = load_b16(kh + (size_t)t0 * DH + 32,        DH, lane);
    v16h B10 = load_b16(kh + (size_t)(t0 + 16) * DH + 0,  DH, lane);
    v16h B11 = load_b16(kh + (size_t)(t0 + 16) * DH + 32, DH, lane);
    v8f c0 = {}, c1 = {};
    c0 = wmma16(A0, B00, c0); c0 = wmma16(A1, B01, c0);
    c1 = wmma16(A0, B10, c1); c1 = wmma16(A1, B11, c1);

    // row max: pack two rows per h2, reduce across the 16-lane group with v_pk_max.
    // (f16 rounding of the offset is harmless: softmax is exactly offset-invariant.)
    float mx[8];
#pragma unroll
    for (int j = 0; j < 4; ++j) {
      h2 hm;
      hm[0] = (_Float16)fmaxf(c0[2 * j],     c1[2 * j]);
      hm[1] = (_Float16)fmaxf(c0[2 * j + 1], c1[2 * j + 1]);
      hm = pkmax(hm, shfl_xor_h2(hm, 1));
      hm = pkmax(hm, shfl_xor_h2(hm, 2));
      hm = pkmax(hm, shfl_xor_h2(hm, 4));
      hm = pkmax(hm, shfl_xor_h2(hm, 8));
      mx[2 * j]     = (float)hm[0];
      mx[2 * j + 1] = (float)hm[1];
    }

    float p0[8], p1[8];
#pragma unroll
    for (int r = 0; r < 8; ++r) {
      float mc = fmaxf(m_i[r], mx[r]);
      float sc = __expf(m_i[r] - mc);
      m_i[r] = mc;
      O0[r] *= sc; O1[r] *= sc; O2[r] *= sc; O3[r] *= sc; O4[r] *= sc;
      p0[r] = __expf(c0[r] - mc);
      p1[r] = __expf(c1[r] - mc);
    }

    // re-layout P: C-tile (f32) -> LDS row-major f16 -> A-tile (16x32)
    _Float16* pb = pbuf[w];
#pragma unroll
    for (int r = 0; r < 8; ++r) {
      pb[(mb + r) * 32 + row]      = (_Float16)p0[r];
      pb[(mb + r) * 32 + 16 + row] = (_Float16)p1[r];
    }
    asm volatile("s_wait_dscnt 0" ::: "memory");  // DS in-order per wave; make ordering explicit
    v16h P;
    {
      const _Float16* rp = pb + row * 32 + (sel << 3);
      v8h lo = *(const v8h*)rp;
      v8h hi = *(const v8h*)(rp + 16);
#pragma unroll
      for (int i = 0; i < 8; ++i) { P[i] = lo[i]; P[i + 8] = hi[i]; }
    }

    O4 = wmma16(P, Bones, O4);  // row sums ride the matrix pipe
    O0 = wmma16(P, load_b16(vh + (size_t)0  * SEQ + t0, SEQ, lane), O0);
    O1 = wmma16(P, load_b16(vh + (size_t)16 * SEQ + t0, SEQ, lane), O1);
    O2 = wmma16(P, load_b16(vh + (size_t)32 * SEQ + t0, SEQ, lane), O2);
    O3 = wmma16(P, load_b16(vh + (size_t)48 * SEQ + t0, SEQ, lane), O3);
  }

#pragma unroll
  for (int r = 0; r < 8; ++r) {
    float inv = 1.0f / O4[r];  // every column of the ones-matmul holds the row sum
    size_t base = (size_t)(s0 + mb + r) * DM + h * DH + row;
    ctx[base + 0]  = (_Float16)(O0[r] * inv);
    ctx[base + 16] = (_Float16)(O1[r] * inv);
    ctx[base + 32] = (_Float16)(O2[r] * inv);
    ctx[base + 48] = (_Float16)(O3[r] * inv);
  }
}

// ---------------- output projection ----------------
// out[s][o] = sum_i ctx[s][i] * Wo[o][i] + bo[o]; Woh rows are B columns (contiguous in K).
__global__ void __launch_bounds__(256) k_oproj(const _Float16* __restrict__ ctx,
                                               const _Float16* __restrict__ Woh,
                                               const float* __restrict__ bo,
                                               float* __restrict__ out) {
  int lane = threadIdx.x & 31;
  int wid  = blockIdx.x * (blockDim.x >> 5) + (threadIdx.x >> 5);  // 0..8191
  int m0 = (wid >> 5) * 16;
  int n0 = (wid & 31) * 16;
  v8f c = {};
#pragma unroll 4
  for (int kk = 0; kk < DM; kk += 32) {
    v16h a = load_a16(ctx + (size_t)m0 * DM + kk, DM, lane);
    v16h b = load_b16(Woh + (size_t)n0 * DM + kk, DM, lane);
    c = wmma16(a, b, c);
  }
  int   n  = n0 + (lane & 15);
  float bn = bo[n];
  int   mb = m0 + ((lane >> 4) << 3);
#pragma unroll
  for (int r = 0; r < 8; ++r) out[(size_t)(mb + r) * DM + n] = c[r] + bn;
}

extern "C" void kernel_launch(void* const* d_in, const int* in_sizes, int n_in,
                              void* d_out, int out_size, void* d_ws, size_t ws_size,
                              hipStream_t stream) {
  const float* x  = (const float*)d_in[0];
  const float* Wq = (const float*)d_in[1];
  const float* Wk = (const float*)d_in[2];
  const float* Wv = (const float*)d_in[3];
  const float* bq = (const float*)d_in[4];
  const float* bk = (const float*)d_in[5];
  const float* bv = (const float*)d_in[6];
  const float* Wo = (const float*)d_in[7];
  const float* bo = (const float*)d_in[8];
  float* out = (float*)d_out;

  // workspace layout (f16 elements)
  _Float16* ws  = (_Float16*)d_ws;
  const size_t N_X  = (size_t)SEQ * DM;        // 2M
  const size_t N_W  = (size_t)NH * DM * DH;    // 256K per matrix
  const size_t N_QK = (size_t)NH * SEQ * DH;   // 2M per tensor
  _Float16* xh  = ws;
  _Float16* WT  = xh + N_X;          // [3][NH][DH][DM]
  _Float16* Woh = WT + 3 * N_W;      // [DM][DM]
  _Float16* qh  = Woh + (size_t)DM * DM;
  _Float16* kh  = qh + N_QK;
  _Float16* vTh = kh + N_QK;         // [NH][DH][SEQ]
  _Float16* ctx = vTh + N_QK;        // [SEQ][DM]

  // prep
  k_cvt<<<(int)(N_X / 256), 256, 0, stream>>>(x, xh, (int)N_X);
  k_transpose_w<<<(int)(N_W / 256), 256, 0, stream>>>(Wq, WT + 0 * N_W);
  k_transpose_w<<<(int)(N_W / 256), 256, 0, stream>>>(Wk, WT + 1 * N_W);
  k_transpose_w<<<(int)(N_W / 256), 256, 0, stream>>>(Wv, WT + 2 * N_W);
  k_cvt<<<(DM * DM) / 256, 256, 0, stream>>>(Wo, Woh, DM * DM);

  // QKV projection: 24576 waves, 8 per block
  k_qkv<<<3072, 256, 0, stream>>>(xh, WT, bq, bk, bv, qh, kh, vTh);

  // flash attention: (SEQ/128) x NH blocks, 8 waves each (one head per block)
  k_attn<<<dim3(SEQ / 128, NH), 256, 0, stream>>>(qh, kh, vTh, ctx);

  // output projection: 8192 waves, 8 per block
  k_oproj<<<1024, 256, 0, stream>>>(ctx, Woh, bo, out);
}
